// MLARoPEKVCacheCatTestModel_84353157693691
// MI455X (gfx1250) — matrix operand order repaired and checked by reading, not compiled, not run on hardware
//
#include <hip/hip_runtime.h>
#include <hip/hip_bf16.h>

// ---------------- problem constants ----------------
#define T_TOK   32768
#define Q_LORA  1536
#define KV_LORA 512
#define ROPE_D  64
#define NOPE_D  128
#define NHEADS  16
#define QKHD    192                 // NOPE_D + ROPE_D
#define HEADSZ  576                 // KV_LORA + ROPE_D
#define NDIM    (NHEADS * QKHD)     // 3072
#define KDIM    Q_LORA              // 1536
#define LROW    (Q_LORA + KV_LORA + ROPE_D)  // 2112 floats per qkv_lora row

// ---------------- WMMA vector types ----------------
typedef __attribute__((ext_vector_type(16))) __bf16 v16bf;
typedef __attribute__((ext_vector_type(8)))  __bf16 v8bf;
typedef __attribute__((ext_vector_type(8)))  float  v8f;

union BFrag { v16bf v; v8bf h[2]; };

// ---------------- tiling ----------------
constexpr int BM = 128;
constexpr int BN = 128;
constexpr int BK = 32;
constexpr int KTILES = KDIM / BK;   // 48
constexpr int APITCH = 40;          // 32 bf16 + 8 pad (80B rows, 16B aligned)
constexpr int BPITCH = 40;

// workspace layout (bytes): [0, 9437184) bf16 Wq ; then cos table ; then sin table
constexpr size_t WS_WQ_ELEMS  = (size_t)KDIM * NDIM;            // bf16
constexpr size_t WS_CS_OFF    = WS_WQ_ELEMS * 2;                 // bytes
constexpr size_t WS_TBL_ELEMS = (size_t)T_TOK * 32;              // floats each

// =====================================================================
// Kernel 1: Wq (f32 row-major) -> bf16 in ws, tiled [kt][n][32 k] so each
// GEMM B tile is one contiguous 8KB block.
// =====================================================================
__global__ __launch_bounds__(256) void pack_wq(const float* __restrict__ Wq,
                                               __bf16* __restrict__ Bws) {
    const int tid = blockIdx.x * 256 + threadIdx.x;   // tid = kt*NDIM + n
    const int n  = tid % NDIM;
    const int kt = tid / NDIM;
    alignas(16) __bf16 tmp[BK];
#pragma unroll
    for (int k = 0; k < BK; ++k)
        tmp[k] = (__bf16)Wq[(size_t)(kt * BK + k) * NDIM + n];
    uint4* dst = (uint4*)(Bws + (size_t)tid * BK);
    const uint4* s = (const uint4*)tmp;
    dst[0] = s[0]; dst[1] = s[1]; dst[2] = s[2]; dst[3] = s[3];
}

// =====================================================================
// Kernel 2: cos/sin table, T x 32 each (8 MB total, L2-resident).
// =====================================================================
__global__ __launch_bounds__(256) void rope_table(const int* __restrict__ positions,
                                                  float* __restrict__ cst,
                                                  float* __restrict__ snt) {
    const int t   = blockIdx.x * 256 + threadIdx.x;   // [0, T*32)
    const int row = t >> 5;
    const int f   = t & 31;
    const float pos = (float)positions[row];
    const float inv = __expf(-((float)f / 32.0f) * 9.210340371976184f); // 10000^(-f/32)
    const float a = pos * inv;
    cst[t] = cosf(a);
    snt[t] = sinf(a);
}

// =====================================================================
// Kernel 3: q = q_c @ Wq via v_wmma_f32_16x16x32_bf16, with NeoX RoPE
// fused into the epilogue (each 64-col wave stripe is either entirely a
// head's rope block or entirely nope).
// Grid: x = N tiles (24, fastest -> A stripe L2 reuse), y = M tiles (256).
// =====================================================================
__global__ __launch_bounds__(256) void gemm_bf16(const float* __restrict__ qkv,
                                                 const __bf16* __restrict__ Bws,
                                                 const float* __restrict__ cst,
                                                 const float* __restrict__ snt,
                                                 float* __restrict__ qout) {
    __shared__ __bf16 Alds[BM * APITCH];
    __shared__ __bf16 Blds[BN * BPITCH];

    const int t    = threadIdx.x;
    const int lane = t & 31;
    const int wid  = t >> 5;
    const int wM   = wid & 3;    // 0..3 -> 32-row stripes
    const int wN   = wid >> 2;   // 0..1 -> 64-col stripes
    const int n0   = blockIdx.x * BN;
    const int m0   = blockIdx.y * BM;

    v8f acc[2][4];
#pragma unroll
    for (int i = 0; i < 2; ++i)
#pragma unroll
        for (int j = 0; j < 4; ++j) acc[i][j] = (v8f)(0.0f);

    const int rsel = lane & 15;
    const bool hi  = lane >= 16;

    // ---- per-thread staging addresses ----
    const int aR  = (t >> 3) + ((t & 256) ? 0 : 0); // placeholder (unused)
    (void)aR;

    // global->register prefetch buffers
    float4 aReg[4];
    uint4  bReg[2];

    auto load_tile = [&](int kt) {
#pragma unroll
        for (int i = 0; i < 4; ++i) {
            const int idx = i * 256 + t;
            const int r   = idx >> 3;
            const int c4  = (idx & 7) << 2;
            aReg[i] = *(const float4*)(qkv + (size_t)(m0 + r) * LROW + kt * BK + c4);
        }
        const uint4* src = (const uint4*)(Bws + ((size_t)kt * NDIM + n0) * BK);
        bReg[0] = src[t * 2 + 0];
        bReg[1] = src[t * 2 + 1];
    };

    load_tile(0);

    for (int kt = 0; kt < KTILES; ++kt) {
        // ---- commit prefetched tile to LDS ----
#pragma unroll
        for (int i = 0; i < 4; ++i) {
            const int idx = i * 256 + t;
            const int r   = idx >> 3;
            const int c4  = (idx & 7) << 2;
            __bf16* d = &Alds[r * APITCH + c4];
            d[0] = (__bf16)aReg[i].x; d[1] = (__bf16)aReg[i].y;
            d[2] = (__bf16)aReg[i].z; d[3] = (__bf16)aReg[i].w;
        }
        {
            const int c = t >> 1, half = t & 1;
            uint4* d = (uint4*)&Blds[c * BPITCH + half * 16];
            d[0] = bReg[0]; d[1] = bReg[1];
        }
        __syncthreads();

        // ---- prefetch next tile (overlaps with WMMA below) ----
        if (kt + 1 < KTILES) load_tile(kt + 1);

        // ---- build fragments (ISA 7.12.2 layouts) ----
        BFrag a[2], b[4];
#pragma unroll
        for (int i = 0; i < 2; ++i) {
            const __bf16* p = &Alds[(wM * 32 + i * 16 + rsel) * APITCH + (hi ? 8 : 0)];
            a[i].h[0] = *(const v8bf*)p;          // K 0-7   (lo) / 8-15  (hi)
            a[i].h[1] = *(const v8bf*)(p + 16);   // K 16-23 (lo) / 24-31 (hi)
        }
#pragma unroll
        for (int j = 0; j < 4; ++j) {
            const __bf16* p = &Blds[(wN * 64 + j * 16 + rsel) * BPITCH + (hi ? 8 : 0)];
            b[j].h[0] = *(const v8bf*)p;          // K 0-7   (lo) / 8-15  (hi)
            b[j].h[1] = *(const v8bf*)(p + 16);   // K 16-23 (lo) / 24-31 (hi)
        }
#pragma unroll
        for (int i = 0; i < 2; ++i)
#pragma unroll
            for (int j = 0; j < 4; ++j)
                acc[i][j] = __builtin_amdgcn_wmma_f32_16x16x32_bf16(
                    false, a[i].v, false, b[j].v, (short)0, acc[i][j], false, false);
        __syncthreads();
    }

    // ---- fused NeoX RoPE on rope stripes (colBase % 192 == 128) ----
    const int nlo = lane & 15;
    const int mof = hi ? 8 : 0;
    const int colBase = n0 + wN * 64;
    if ((colBase % QKHD) == NOPE_D) {
#pragma unroll
        for (int i = 0; i < 2; ++i)
#pragma unroll
            for (int r = 0; r < 8; ++r) {
                const int row = m0 + wM * 32 + i * 16 + r + mof;
#pragma unroll
                for (int j = 0; j < 2; ++j) {
                    const int f = 16 * j + nlo;               // rope freq index 0..31
                    const float c = cst[(size_t)row * 32 + f];
                    const float s = snt[(size_t)row * 32 + f];
                    const float x1 = acc[i][j][r];
                    const float x2 = acc[i][j + 2][r];
                    acc[i][j][r]     = x1 * c - x2 * s;
                    acc[i][j + 2][r] = x2 * c + x1 * s;
                }
            }
    }

    // ---- store q (write-once stream -> non-temporal) ----
#pragma unroll
    for (int i = 0; i < 2; ++i)
#pragma unroll
        for (int j = 0; j < 4; ++j)
#pragma unroll
            for (int r = 0; r < 8; ++r) {
                const int row = m0 + wM * 32 + i * 16 + r + mof;
                const int col = colBase + j * 16 + nlo;
                __builtin_nontemporal_store(acc[i][j][r],
                                            qout + (size_t)row * NDIM + col);
            }
}

// =====================================================================
// Kernel 4: kv_c copy, k_pe RoPE (table-driven), kv_cache scatter.
// =====================================================================
__global__ __launch_bounds__(256) void kv_epilogue(const float* __restrict__ qkv,
                                                   const float* __restrict__ cst,
                                                   const float* __restrict__ snt,
                                                   const int* __restrict__ slot_mapping,
                                                   float* __restrict__ out) {
    const int row = blockIdx.x;
    float* kv_c_out  = out + (size_t)T_TOK * NDIM;            // T x 512
    float* k_pe_out  = kv_c_out + (size_t)T_TOK * KV_LORA;    // T x 64
    float* cache_out = k_pe_out + (size_t)T_TOK * ROPE_D;     // T x 576

    const int t = threadIdx.x;
    const float* in = qkv + (size_t)row * LROW;
    const int slot  = slot_mapping[row];

    // kv_c: copy to output slice and cache entry
    for (int c = t; c < KV_LORA; c += 256) {
        const float v = in[Q_LORA + c];
        __builtin_nontemporal_store(v, kv_c_out + (size_t)row * KV_LORA + c);
        __builtin_nontemporal_store(v, cache_out + (size_t)slot * HEADSZ + c);
    }

    // k_pe RoPE: 32 pairs
    if (t < 32) {
        const int j = t;
        const float c = cst[(size_t)row * 32 + j];
        const float s = snt[(size_t)row * 32 + j];
        const float x1 = in[Q_LORA + KV_LORA + j];
        const float x2 = in[Q_LORA + KV_LORA + 32 + j];
        const float o1 = x1 * c - x2 * s;
        const float o2 = x2 * c + x1 * s;
        __builtin_nontemporal_store(o1, k_pe_out + (size_t)row * ROPE_D + j);
        __builtin_nontemporal_store(o2, k_pe_out + (size_t)row * ROPE_D + 32 + j);
        __builtin_nontemporal_store(o1, cache_out + (size_t)slot * HEADSZ + KV_LORA + j);
        __builtin_nontemporal_store(o2, cache_out + (size_t)slot * HEADSZ + KV_LORA + 32 + j);
    }
}

// =====================================================================
extern "C" void kernel_launch(void* const* d_in, const int* in_sizes, int n_in,
                              void* d_out, int out_size, void* d_ws, size_t ws_size,
                              hipStream_t stream) {
    const float* qkv          = (const float*)d_in[0];
    const int*   positions    = (const int*)d_in[1];
    const float* Wq           = (const float*)d_in[2];
    const float* kv_cache_in  = (const float*)d_in[3];
    const int*   slot_mapping = (const int*)d_in[4];
    float*  out = (float*)d_out;

    __bf16* Bws = (__bf16*)d_ws;                              // 9.4 MB
    float*  cst = (float*)((char*)d_ws + WS_CS_OFF);          // 4 MB
    float*  snt = cst + WS_TBL_ELEMS;                         // 4 MB

    // 1) Wq -> bf16, tiled for contiguous B-tile fetches
    pack_wq<<<KTILES * NDIM / 256, 256, 0, stream>>>(Wq, Bws);

    // 2) RoPE cos/sin table
    rope_table<<<(T_TOK * 32) / 256, 256, 0, stream>>>(positions, cst, snt);

    // 3) q = q_c @ Wq (bf16 WMMA, f32 acc) with fused NeoX RoPE
    dim3 g(NDIM / BN, T_TOK / BM);   // x = N (fast -> A stripe reuse in L2)
    gemm_bf16<<<g, 256, 0, stream>>>(qkv, Bws, cst, snt, out);

    // 4) seed the cache output region with the input cache (graph-safe async d2d)
    float* cache_out = out + (size_t)T_TOK * NDIM + (size_t)T_TOK * KV_LORA + (size_t)T_TOK * ROPE_D;
    hipMemcpyAsync(cache_out, kv_cache_in, (size_t)T_TOK * HEADSZ * sizeof(float),
                   hipMemcpyDeviceToDevice, stream);

    // 5) kv_c / k_pe / cache scatter
    kv_epilogue<<<T_TOK, 256, 0, stream>>>(qkv, cst, snt, slot_mapping, out);
}
